// GATConv_26834955665707
// MI455X (gfx1250) — matrix operand order repaired and checked by reading, not compiled, not run on hardware
//
#include <hip/hip_runtime.h>
#include <hip/hip_bf16.h>

typedef __attribute__((ext_vector_type(16))) _Float16 v16h;
typedef __attribute__((ext_vector_type(8)))  float    v8f;

#define IN_F   128
#define HEADS  4
#define DOUT   32
#define NEG_SLOPE 0.2f

// ---------- order-preserving float <-> uint mapping for atomicMax ----------
__device__ __forceinline__ unsigned flipf(float f) {
    unsigned u = __float_as_uint(f);
    return (u & 0x80000000u) ? ~u : (u | 0x80000000u);
}
__device__ __forceinline__ float unflipf(unsigned u) {
    return (u & 0x80000000u) ? __uint_as_float(u & 0x7FFFFFFFu)
                             : __uint_as_float(~u);
}

// ---------- 0: init out, esum, emax keys ----------
__global__ void gat_init(float* __restrict__ out, float* __restrict__ esum,
                         unsigned* __restrict__ emax, int N) {
    int t = blockIdx.x * blockDim.x + threadIdx.x;
    int n128 = N * IN_F;
    if (t < n128) out[t] = 0.0f;
    if (t < N * HEADS) { esum[t] = 0.0f; emax[t] = 0u; }  // key 0 < flip(-FLT_MAX)
}

// ---------- 1: ft = feat @ W^T via WMMA f16 hi/lo split (fp32-accurate) ----
// one wave per 16x16 output tile; block = 8 waves; blockIdx.x = tileM, wave = tileN
__global__ void gat_gemm_wmma(const float* __restrict__ feat,
                              const float* __restrict__ W,
                              float* __restrict__ ft) {
    const int lane  = threadIdx.x & 31;
    const int tileN = threadIdx.x >> 5;       // 0..7  (128 output cols / 16)
    const int tileM = blockIdx.x;             // N/16 tiles
    const int half  = lane >> 4;              // 0/1
    const int rc    = lane & 15;              // A row (M) / B col (N) within tile

    const float* __restrict__ arow = feat + (size_t)(tileM * 16 + rc) * IN_F;
    const float* __restrict__ brow = W    + (size_t)(tileN * 16 + rc) * IN_F; // W[o][k] -> B[k][o]

    v8f c = {};
    for (int k0 = 0; k0 < IN_F; k0 += 32) {
        v16h ahi, alo, bhi, blo;
        // A 16x32 layout: elems 0..7 -> K = k0 + half*8 + j ; 8..15 -> K = k0+16+half*8+j
#pragma unroll
        for (int j = 0; j < 8; ++j) {
            float a0 = arow[k0 + half * 8 + j];
            _Float16 h0 = (_Float16)a0;
            ahi[j] = h0;
            alo[j] = (_Float16)(a0 - (float)h0);
            float a1 = arow[k0 + 16 + half * 8 + j];
            _Float16 h1 = (_Float16)a1;
            ahi[j + 8] = h1;
            alo[j + 8] = (_Float16)(a1 - (float)h1);
        }
        // B 32x16 layout: lane holds col N=rc, elems 0..7 -> K = k0 + half*8 + j ; 8..15 -> K = k0+16+half*8+j
#pragma unroll
        for (int j = 0; j < 16; ++j) {
            float b0 = brow[k0 + (j >> 3) * 16 + half * 8 + (j & 7)];
            _Float16 hb = (_Float16)b0;
            bhi[j] = hb;
            blo[j] = (_Float16)(b0 - (float)hb);
        }
        // fp32 emulation: a*b ~= ahi*bhi + ahi*blo + alo*bhi (f32 accumulate)
        c = __builtin_amdgcn_wmma_f32_16x16x32_f16(false, ahi, false, bhi, (short)0, c, false, false);
        c = __builtin_amdgcn_wmma_f32_16x16x32_f16(false, ahi, false, blo, (short)0, c, false, false);
        c = __builtin_amdgcn_wmma_f32_16x16x32_f16(false, alo, false, bhi, (short)0, c, false, false);
    }
    // C/D layout: vgpr r, lane l -> D[M = r + 8*half][N = l&15]
    float* crow = ft + (size_t)tileM * 16 * IN_F + tileN * 16 + rc;
#pragma unroll
    for (int r = 0; r < 8; ++r)
        crow[(size_t)(r + 8 * half) * IN_F] = c[r];
}

// ---------- 2: el/er per (node, head): wave reduction over D=32 ----------
__global__ void gat_elr(const float* __restrict__ ft,
                        const float* __restrict__ attn_l,
                        const float* __restrict__ attn_r,
                        float* __restrict__ el, float* __restrict__ er, int NH) {
    int wid  = (blockIdx.x * blockDim.x + threadIdx.x) >> 5;  // (n,h) index
    int lane = threadIdx.x & 31;                              // d
    if (wid >= NH) return;
    int h = wid & 3;
    float v  = ft[(size_t)wid * DOUT + lane];                 // wid*32 == n*128 + h*32
    float sl = v * attn_l[h * DOUT + lane];
    float sr = v * attn_r[h * DOUT + lane];
#pragma unroll
    for (int off = 16; off > 0; off >>= 1) {
        sl += __shfl_down(sl, off, 32);
        sr += __shfl_down(sr, off, 32);
    }
    if (lane == 0) { el[wid] = sl; er[wid] = sr; }
}

// ---------- 3: segment max over edges (thread per edge*head) ----------
__global__ void gat_edge_max(const int* __restrict__ src, const int* __restrict__ dst,
                             const float* __restrict__ el, const float* __restrict__ er,
                             unsigned* __restrict__ emax, int E4) {
    int t = blockIdx.x * blockDim.x + threadIdx.x;
    if (t >= E4) return;
    int e = t >> 2, h = t & 3;
    int s = src[e], d = dst[e];
    float v = el[s * HEADS + h] + er[d * HEADS + h];
    v = v > 0.0f ? v : NEG_SLOPE * v;
    atomicMax(&emax[d * HEADS + h], flipf(v));
}

// ---------- 4: exp + unnormalized weighted scatter (wave per edge) ----------
__global__ void gat_edge_accum(const int* __restrict__ src, const int* __restrict__ dst,
                               const float* __restrict__ el, const float* __restrict__ er,
                               const unsigned* __restrict__ emax,
                               const float* __restrict__ ft,
                               float* __restrict__ esum, float* __restrict__ out, int E) {
    int wid  = (blockIdx.x * blockDim.x + threadIdx.x) >> 5;
    int lane = threadIdx.x & 31;
    if (wid >= E) return;
    int s = src[wid], d = dst[wid];
    int h = lane & 3;
    float v = el[s * HEADS + h] + er[d * HEADS + h];
    v = v > 0.0f ? v : NEG_SLOPE * v;
    float ez = __expf(v - unflipf(emax[d * HEADS + h]));
    if (lane < HEADS) atomicAdd(&esum[d * HEADS + lane], ez);  // lane==h here
    const float* fs = ft + (size_t)s * IN_F;
    float* od = out + (size_t)d * IN_F;
#pragma unroll
    for (int hh = 0; hh < HEADS; ++hh) {
        float ezh = __shfl(ez, hh, 32);      // lane hh has head hh's weight
        int idx = hh * DOUT + lane;          // lane = feature d
        atomicAdd(&od[idx], ezh * fs[idx]);
    }
}

// ---------- 5: normalize by segment sum ----------
__global__ void gat_norm(float* __restrict__ out, const float* __restrict__ esum, int n128) {
    int t = blockIdx.x * blockDim.x + threadIdx.x;
    if (t >= n128) return;
    int n = t >> 7, h = (t >> 5) & 3;
    float s = esum[n * HEADS + h];
    out[t] = (s > 0.0f) ? out[t] / s : 0.0f;
}

extern "C" void kernel_launch(void* const* d_in, const int* in_sizes, int n_in,
                              void* d_out, int out_size, void* d_ws, size_t ws_size,
                              hipStream_t stream) {
    const float* feat   = (const float*)d_in[0];
    const float* W      = (const float*)d_in[1];
    const float* attn_l = (const float*)d_in[2];
    const float* attn_r = (const float*)d_in[3];
    const int*   src    = (const int*)d_in[4];
    const int*   dst    = (const int*)d_in[5];
    float* out = (float*)d_out;

    const int N = in_sizes[0] / IN_F;   // 100000
    const int E = in_sizes[4];          // 1600000

    // workspace layout
    float*    ft   = (float*)d_ws;                    // N*128
    float*    el   = ft   + (size_t)N * IN_F;         // N*4
    float*    er   = el   + (size_t)N * HEADS;        // N*4
    float*    esum = er   + (size_t)N * HEADS;        // N*4
    unsigned* emax = (unsigned*)(esum + (size_t)N * HEADS); // N*4

    const int n128 = N * IN_F;
    gat_init<<<(n128 + 255) / 256, 256, 0, stream>>>(out, esum, emax, N);

    // GEMM: one M-tile per block (N/16 blocks), 8 waves = 8 N-tiles (128 cols)
    gat_gemm_wmma<<<N / 16, 256, 0, stream>>>(feat, W, ft);

    const int NH = N * HEADS;
    gat_elr<<<(NH + 7) / 8, 256, 0, stream>>>(ft, attn_l, attn_r, el, er, NH);

    gat_edge_max<<<(E * HEADS + 255) / 256, 256, 0, stream>>>(src, dst, el, er, emax, E * HEADS);

    gat_edge_accum<<<(E + 7) / 8, 256, 0, stream>>>(src, dst, el, er, emax, ft, esum, out, E);

    gat_norm<<<(n128 + 255) / 256, 256, 0, stream>>>(out, esum, n128);
}
